// PBL_51754355917022
// MI455X (gfx1250) — matrix operand order repaired and checked by reading, not compiled, not run on hardware
//
#include <hip/hip_runtime.h>
#include <hip/hip_bf16.h>

typedef __attribute__((ext_vector_type(16))) _Float16 v16h;
typedef __attribute__((ext_vector_type(8)))  _Float16 v8h;
typedef __attribute__((ext_vector_type(8)))  float    v8f;

// Global-address-space pointer types (keep AS info through asm laundering so
// B loads lower to global_load_b128, not flat_load_b128 which also ties up
// the LDS/DScnt path).
typedef const _Float16 __attribute__((address_space(1)))* ghptr;
typedef const v16h     __attribute__((address_space(1)))* gv16ptr;

// Problem constants (from reference setup_inputs)
#define NSEQ   1024
#define TLEN   128
#define HID    256
#define NACT   18
#define MDIM   512
#define SSUB   6
#define FSUB   2
#define KHOR   20
#define BATCH  (NSEQ * SSUB)        // 6144
#define FB     (FSUB * BATCH)       // 12288
#define GROWS  32                   // batch rows per persistent GRU block

// ---------------------------------------------------------------------------
// Pack a weight matrix into the CDNA5 WMMA f16 B-operand per-lane layout.
// Logical B is K x N (C = A[MxK] * B[KxN]).
//   transposed==1: src is [N x K] row-major (W_hh: gh = h @ W_hh^T)
//   transposed==0: src is [K x N] row-major (W1/W2/W3: x @ W)
// Packed layout: [ntile][kchunk][lane 0..31][16 halves], lane l holds
// column n = ntile*16 + (l&15), K = kchunk*32 + (l>>4)*8 + {0..7, 16..23}.
// ---------------------------------------------------------------------------
__global__ void pack_b_f16(const float* __restrict__ src, _Float16* __restrict__ dst,
                           int K, int Nc, int transposed) {
    int i = blockIdx.x * 256 + threadIdx.x;
    if (i >= K * Nc) return;
    int hh   = i & 15;
    int lane = (i >> 4) & 31;
    int rem  = i >> 9;
    int kcCnt = K >> 5;
    int kc = rem % kcCnt;
    int nt = rem / kcCnt;
    int n = nt * 16 + (lane & 15);
    int k = kc * 32 + ((lane >> 4) << 3) + (hh & 7) + ((hh >> 3) << 4);
    float v = transposed ? src[(size_t)n * K + k] : src[(size_t)k * Nc + n];
    dst[i] = (_Float16)v;
}

// h0[b, :] = b_t[n, time_sub[s], :]   with b = n*SSUB + s
__global__ void prep_h0(const float* __restrict__ b_t, const int* __restrict__ tsub,
                        float* __restrict__ h0) {
    int idx = blockIdx.x * 256 + threadIdx.x;
    if (idx >= BATCH * HID) return;
    int b = idx / HID, c = idx - b * HID;
    int n = b / SSUB, s = b - n * SSUB;
    int t = tsub[s];
    h0[idx] = b_t[((size_t)n * TLEN + t) * HID + c];
}

// aidx[step, b] = actions[n, tsub[s]+step] if in range of the (T-1)-slice, else -1
__global__ void prep_aidx(const int* __restrict__ actions, const int* __restrict__ tsub,
                          int* __restrict__ aidx) {
    int idx = blockIdx.x * 256 + threadIdx.x;
    if (idx >= KHOR * BATCH) return;
    int step = idx / BATCH, b = idx - step * BATCH;
    int n = b / SSUB, s = b - n * SSUB;
    int t = tsub[s] + step;
    aidx[idx] = (t <= TLEN - 2) ? actions[n * TLEN + t] : -1;
}

__global__ void zero_acc(float* acc) {
    if (threadIdx.x < 4) acc[threadIdx.x] = 0.f;
}

__device__ __forceinline__ v16h make_a16(v8h p0, v8h p1) {
    return __builtin_shufflevector(p0, p1, 0,1,2,3,4,5,6,7,8,9,10,11,12,13,14,15);
}

#define WMMA_F16(A, B, C) \
    __builtin_amdgcn_wmma_f32_16x16x32_f16(false, (A), false, (B), (short)0, (C), false, false)

// ---------------------------------------------------------------------------
// Persistent GRU rollout: one block owns GROWS=32 batch rows for all 20
// steps. f16 A-tile (h) lives in LDS and is rewritten in place each step;
// fp32 h lives in registers (each wave owns one 16-column stripe).
// Block: 512 threads = 16 waves; wave w handles column tile jt=w (16 cols)
// for both row tiles, 3 gates => 48 WMMAs per step. Gate-weight B fragments
// are re-streamed from L2 each step through a laundered global-AS pointer:
// the opaque pointer stops LLVM hoisting ~192 VGPRs of loop-invariant loads
// (round-2 scratch spills), and the explicit AS(1) keeps them on the
// global_load path (LOADcnt only) instead of flat_load (LOADcnt + DScnt).
// Grid: BATCH/GROWS = 192 blocks.
// ---------------------------------------------------------------------------
__global__ void __launch_bounds__(512)
gru_rollout(const float* __restrict__ h0, const _Float16* __restrict__ Bw,
            const float* __restrict__ Wih, const float* __restrict__ bih,
            const float* __restrict__ bhh,
            const int* __restrict__ aidx, const int* __restrict__ unroll,
            float* __restrict__ hpred) {
    __shared__ alignas(16) _Float16 Ah[GROWS * HID];   // 16 KB f16 h tile
    __shared__ int aL[KHOR * GROWS];                   // 2.5 KB action ids
    const int tid = threadIdx.x;
    const int rowBase = blockIdx.x * GROWS;

    // Stage initial f16 A tile and all per-step action ids
    for (int idx = tid; idx < GROWS * HID; idx += 512) {
        int r = idx >> 8, c = idx & (HID - 1);
        Ah[idx] = (_Float16)h0[(size_t)(rowBase + r) * HID + c];
    }
    for (int idx = tid; idx < KHOR * GROWS; idx += 512) {
        int st = idx >> 5, r = idx & (GROWS - 1);
        aL[idx] = aidx[st * BATCH + rowBase + r];
    }

    const int lane = tid & 31;
    const int jt = tid >> 5;                 // wave id = column tile, 0..15
    const int jn = jt * 16 + (lane & 15);    // hidden unit (C/D col = lane&15)
    const int arow = lane & 15;
    const int asel = (lane >> 4) << 3;       // A-layout half select (0 or 8)
    const int hisel = (lane >> 4) << 3;      // C/D row-half select (0 or 8)
    const int u0 = unroll[0], u1 = unroll[1];
    const float bir = bih[jn], biz = bih[jn + HID], bin = bih[jn + 2 * HID];
    const float bhr = bhh[jn], bhz = bhh[jn + HID], bhn = bhh[jn + 2 * HID];

    // Register-resident fp32 h for this wave's output slots
    float hreg[2][8];
#pragma unroll
    for (int rt = 0; rt < 2; ++rt)
#pragma unroll
        for (int v = 0; v < 8; ++v)
            hreg[rt][v] = h0[(size_t)(rowBase + rt * 16 + v + hisel) * HID + jn];

    __syncthreads();

    const ghptr Bg = (ghptr)Bw;
    for (int step = 0; step < KHOR; ++step) {
        // Launder the weight pointer (opaque per iteration): B-fragment loads
        // stay inside the loop as per-step L2 hits instead of being hoisted
        // and spilled. AS(1) keeps them as global_load_b128 (saddr form).
        ghptr bp = Bg;
        asm volatile("" : "+s"(bp));

        v8f a0r = {}, a0z = {}, a0n = {}, a1r = {}, a1z = {}, a1n = {};
#pragma unroll
        for (int kc = 0; kc < HID / 32; ++kc) {
            // gate weight fragments for this column tile (reused for both row tiles)
            v16h b0 = *(gv16ptr)(bp + ((size_t)(( 0 + jt) * 8 + kc) * 32 + lane) * 16);
            v16h b1 = *(gv16ptr)(bp + ((size_t)((16 + jt) * 8 + kc) * 32 + lane) * 16);
            v16h b2 = *(gv16ptr)(bp + ((size_t)((32 + jt) * 8 + kc) * 32 + lane) * 16);
            v8h p00 = *(const v8h*)(Ah + arow * HID + kc * 32 + asel);
            v8h p01 = *(const v8h*)(Ah + arow * HID + kc * 32 + 16 + asel);
            v16h aF0 = make_a16(p00, p01);
            v8h p10 = *(const v8h*)(Ah + (16 + arow) * HID + kc * 32 + asel);
            v8h p11 = *(const v8h*)(Ah + (16 + arow) * HID + kc * 32 + 16 + asel);
            v16h aF1 = make_a16(p10, p11);
            a0r = WMMA_F16(aF0, b0, a0r);
            a0z = WMMA_F16(aF0, b1, a0z);
            a0n = WMMA_F16(aF0, b2, a0n);
            a1r = WMMA_F16(aF1, b0, a1r);
            a1z = WMMA_F16(aF1, b1, a1z);
            a1n = WMMA_F16(aF1, b2, a1n);
        }
        __syncthreads();   // all A-tile reads complete before in-place rewrite

#pragma unroll
        for (int rt = 0; rt < 2; ++rt) {
#pragma unroll
            for (int v = 0; v < 8; ++v) {
                int ml = rt * 16 + v + hisel;            // local row (C/D layout)
                int act = aL[step * GROWS + ml];
                float gir = bir, giz = biz, gin = bin;
                if (act >= 0) {
                    gir += Wih[(size_t)jn * NACT + act];
                    giz += Wih[(size_t)(jn + HID) * NACT + act];
                    gin += Wih[(size_t)(jn + 2 * HID) * NACT + act];
                }
                float ghr = (rt ? a1r[v] : a0r[v]) + bhr;
                float ghz = (rt ? a1z[v] : a0z[v]) + bhz;
                float ghn = (rt ? a1n[v] : a0n[v]) + bhn;
                float r = 1.f / (1.f + __expf(-(gir + ghr)));
                float z = 1.f / (1.f + __expf(-(giz + ghz)));
                float nn = tanhf(gin + r * ghn);
                float hn = (1.f - z) * nn + z * hreg[rt][v];
                hreg[rt][v] = hn;
                Ah[ml * HID + jn] = (_Float16)hn;        // next step's A operand
                if (step == u0) hpred[(size_t)(rowBase + ml) * HID + jn] = hn;
                if (step == u1) hpred[(size_t)(BATCH + rowBase + ml) * HID + jn] = hn;
            }
        }
        __syncthreads();   // new A tile visible before next step's reads
    }
}

// ---------------------------------------------------------------------------
// Generic WMMA GEMM: C[M,Nc] = act(A[M,K] @ Bpacked + bias), fp32 in/out.
// Block: 256 threads = 8 waves; tile 16 rows x 128 cols. Grid (M/16, Nc/128).
// ---------------------------------------------------------------------------
__global__ void gemm_wmma(const float* __restrict__ Asrc, const _Float16* __restrict__ Bp,
                          const float* __restrict__ bias, float* __restrict__ C,
                          int K, int Nc, int relu) {
    __shared__ alignas(16) _Float16 Ah[16 * 512];    // up to 16 KB
    const int tid = threadIdx.x;
    const int rowBase = blockIdx.x * 16;
    const int total = 16 * K;
    for (int idx = tid; idx < total; idx += 256) {
        int r = idx / K, c = idx - r * K;
        Ah[idx] = (_Float16)Asrc[(size_t)(rowBase + r) * K + c];
    }
    __syncthreads();

    const int lane = tid & 31, w = tid >> 5;
    const int nt = blockIdx.y * 8 + w;
    const int kcCnt = K >> 5;
    const int arow = lane & 15;
    const int asel = (lane >> 4) << 3;

    v8f acc = {};
    for (int kc = 0; kc < kcCnt; ++kc) {
        v8h p0 = *(const v8h*)(Ah + arow * K + kc * 32 + asel);
        v8h p1 = *(const v8h*)(Ah + arow * K + kc * 32 + 16 + asel);
        v16h a = make_a16(p0, p1);
        v16h b = *(const v16h*)(Bp + ((size_t)(nt * kcCnt + kc) * 32 + lane) * 16);
        acc = WMMA_F16(a, b, acc);
    }
    const int n = nt * 16 + (lane & 15);
    const float bv = bias[n];
#pragma unroll
    for (int v = 0; v < 8; ++v) {
        int m = rowBase + v + ((lane >> 4) << 3);
        float o = acc[v] + bv;
        if (relu) o = fmaxf(o, 0.f);
        C[(size_t)m * Nc + n] = o;
    }
}

__device__ __forceinline__ float wave_sum(float v) {
#pragma unroll
    for (int off = 16; off >= 1; off >>= 1) v += __shfl_xor(v, off, 32);
    return v;
}

// ---------------------------------------------------------------------------
// Loss: one wave per (f,b) row. L2-norm pred, reg term, gather+normalize
// target from z_t, mask from cumprod(1-dones), atomic reduce.
// ---------------------------------------------------------------------------
__global__ void loss_reduce(const float* __restrict__ y3, const float* __restrict__ z_t,
                            const float* __restrict__ dones,
                            const int* __restrict__ tsub, const int* __restrict__ unroll,
                            float* __restrict__ acc) {
    const int lane = threadIdx.x & 31;
    const int r = blockIdx.x * 8 + (threadIdx.x >> 5);
    if (r >= FB) return;
    const int f = r / BATCH, b = r - f * BATCH;
    const int n = b / SSUB, s = b - n * SSUB;
    const int t0 = tsub[s];
    const int ku = unroll[f];

    float px[8];
    float ss = 0.f;
#pragma unroll
    for (int i = 0; i < 8; ++i) {
        int c = i * 32 + lane;
        float v = y3[(size_t)r * HID + c];
        px[i] = v; ss += v * v;
    }
    ss = wave_sum(ss);
    float nrm = sqrtf(ss);
    float reg = 0.02f * (nrm - 1.f) * (nrm - 1.f);
    float inv = 1.f / fmaxf(nrm, 1e-8f);

    int tz = t0 + ku;
    bool valid = (tz <= TLEN - 2);
    float tx[8];
    float st = 0.f;
#pragma unroll
    for (int i = 0; i < 8; ++i) {
        int c = i * 32 + lane;
        float v = valid ? z_t[((size_t)n * TLEN + tz + 1) * HID + c] : 0.f;
        tx[i] = v; st += v * v;
    }
    st = wave_sum(st);
    float invt = 1.f / fmaxf(sqrtf(st), 1e-8f);

    float le = 0.f;
#pragma unroll
    for (int i = 0; i < 8; ++i) {
        float d = px[i] * inv - tx[i] * invt;
        le += d * d;
    }
    le = wave_sum(le) * (1.f / (float)HID);

    float m = 1.f;
    for (int i = 0; i <= ku; ++i) {
        int tt = t0 + i;
        float nd = (tt <= TLEN - 2) ? (1.f - dones[n * TLEN + tt]) : 0.f;
        m *= nd;
    }
    float mask = (m > 0.f) ? 1.f : 0.f;

    if (lane == 0) {
        atomicAdd(&acc[0], le * mask);
        atomicAdd(&acc[1], mask);
        atomicAdd(&acc[2], reg);
    }
}

__global__ void finalize(const float* __restrict__ acc, float* __restrict__ out) {
    out[0] = acc[0] / fmaxf(acc[1], 1.f) + acc[2] * (1.f / (float)FB);
}

// ---------------------------------------------------------------------------
extern "C" void kernel_launch(void* const* d_in, const int* in_sizes, int n_in,
                              void* d_out, int out_size, void* d_ws, size_t ws_size,
                              hipStream_t stream) {
    (void)in_sizes; (void)n_in; (void)out_size; (void)ws_size;
    const int*   actions = (const int*)  d_in[0];
    const float* dones   = (const float*)d_in[1];
    const float* b_t     = (const float*)d_in[2];
    const float* z_t     = (const float*)d_in[3];
    const int*   tsub    = (const int*)  d_in[4];
    const int*   usub    = (const int*)  d_in[5];
    const float* W_ih    = (const float*)d_in[6];
    const float* W_hh    = (const float*)d_in[7];
    const float* b_ih    = (const float*)d_in[8];
    const float* b_hh    = (const float*)d_in[9];
    const float* W1      = (const float*)d_in[10];
    const float* b1      = (const float*)d_in[11];
    const float* W2      = (const float*)d_in[12];
    const float* b2      = (const float*)d_in[13];
    const float* W3      = (const float*)d_in[14];
    const float* b3      = (const float*)d_in[15];
    float* out = (float*)d_out;

    // Workspace carve-up (256B aligned)
    char* base = (char*)d_ws;
    size_t off = 0;
    auto carve = [&](size_t bytes) {
        off = (off + 255) & ~(size_t)255;
        void* p = base + off;
        off += bytes;
        return p;
    };
    _Float16* WhhP = (_Float16*)carve((size_t)768 * 256 * 2);
    _Float16* W1P  = (_Float16*)carve((size_t)256 * 512 * 2);
    _Float16* W2P  = (_Float16*)carve((size_t)512 * 512 * 2);
    _Float16* W3P  = (_Float16*)carve((size_t)512 * 256 * 2);
    float* hA    = (float*)carve((size_t)BATCH * HID * 4);
    float* hpred = (float*)carve((size_t)FB * HID * 4);
    float* y1    = (float*)carve((size_t)FB * MDIM * 4);
    float* y2    = (float*)carve((size_t)FB * MDIM * 4);
    float* y3    = (float*)carve((size_t)FB * HID * 4);
    int*   aidx  = (int*)  carve((size_t)KHOR * BATCH * 4);
    float* acc   = (float*)carve(4 * 4);

    // Pack weights to WMMA B layout (f16)
    pack_b_f16<<<(768 * 256 + 255) / 256, 256, 0, stream>>>(W_hh, WhhP, 256, 768, 1);
    pack_b_f16<<<(256 * 512 + 255) / 256, 256, 0, stream>>>(W1, W1P, 256, 512, 0);
    pack_b_f16<<<(512 * 512 + 255) / 256, 256, 0, stream>>>(W2, W2P, 512, 512, 0);
    pack_b_f16<<<(512 * 256 + 255) / 256, 256, 0, stream>>>(W3, W3P, 512, 256, 0);

    // h0 gather + action index table
    prep_h0<<<(BATCH * HID + 255) / 256, 256, 0, stream>>>(b_t, tsub, hA);
    prep_aidx<<<(KHOR * BATCH + 255) / 256, 256, 0, stream>>>(actions, tsub, aidx);
    zero_acc<<<1, 64, 0, stream>>>(acc);

    // Persistent 20-step GRU rollout: h stays in LDS/registers, only hpred
    // leaves the block.
    gru_rollout<<<BATCH / GROWS, 512, 0, stream>>>(
        hA, WhhP, W_ih, b_ih, b_hh, aidx, usub, hpred);

    // MLP head: 256 -> 512 -> 512 -> 256
    gemm_wmma<<<dim3(FB / 16, MDIM / 128), 256, 0, stream>>>(hpred, W1P, b1, y1, 256, 512, 1);
    gemm_wmma<<<dim3(FB / 16, MDIM / 128), 256, 0, stream>>>(y1,    W2P, b2, y2, 512, 512, 1);
    gemm_wmma<<<dim3(FB / 16, HID  / 128), 256, 0, stream>>>(y2,    W3P, b3, y3, 512, 256, 0);

    // Norms, targets, mask, reduction
    loss_reduce<<<(FB + 7) / 8, 256, 0, stream>>>(y3, z_t, dones, tsub, usub, acc);
    finalize<<<1, 1, 0, stream>>>(acc, out);
}
